// RNNLayer_42185168781704
// MI455X (gfx1250) — matrix operand order repaired and checked by reading, not compiled, not run on hardware
//
#include <hip/hip_runtime.h>

// ---------------------------------------------------------------------------
// LSTM layer for MI455X (gfx1250): bf16 WMMA recurrent GEMM + fused gates.
// B=64, T=128, F=H=1024.  W bf16 (16MB) stays L2-resident across 128 steps.
// Per-step GEMM is the sequential critical path -> split K across wave halves
// (x-projection vs h-projection) and reduce partials through LDS to double
// workgroup parallelism and halve each wave's dependent WMMA chain.
// ---------------------------------------------------------------------------

#define B_    64
#define T_    128
#define F_    1024
#define H_    1024
#define N4H   4096      // 4*H
#define KTOT  2048      // F+H

typedef __attribute__((ext_vector_type(16))) __bf16 v16bf;
typedef __attribute__((ext_vector_type(8)))  float  v8f;

__device__ __forceinline__ unsigned short f32_to_bf16(float f) {
    unsigned int u = __float_as_uint(f);
    unsigned int r = u + 0x7FFFu + ((u >> 16) & 1u);   // round-to-nearest-even
    return (unsigned short)(r >> 16);
}

// B fragment: 32 contiguous bytes per lane (dwords 0..7) -> v16bf
__device__ __forceinline__ v16bf load_bfrag(const unsigned short* p) {
    union { uint4 q[2]; v16bf v; } f;
    f.q[0] = *reinterpret_cast<const uint4*>(p);
    f.q[1] = *reinterpret_cast<const uint4*>(p + 8);
    return f.v;
}

// A fragment (16x32 bf16, row-major source): lane group g = lane>>4.
// v0-3 = K[g*8 .. g*8+7], v4-7 = K[16+g*8 .. 16+g*8+7]  (p already has +g*8)
__device__ __forceinline__ v16bf load_afrag(const unsigned short* p) {
    union { uint4 q[2]; v16bf v; } f;
    f.q[0] = *reinterpret_cast<const uint4*>(p);
    f.q[1] = *reinterpret_cast<const uint4*>(p + 16);
    return f.v;
}

// ---------------------------------------------------------------------------
// One-time conversion passes
// ---------------------------------------------------------------------------
__global__ __launch_bounds__(256) void convert_x_kernel(
    const float* __restrict__ x, unsigned short* __restrict__ xb) {
    size_t tid = (size_t)blockIdx.x * 256 + threadIdx.x;   // B*T*F = 8388608
    xb[tid] = f32_to_bf16(x[tid]);
}

// Pack W [2048, 4096] row-major f32 into bf16 WMMA B-fragments.
// Fragment (nt, kt): 16 N cols x 32 K rows = 1024 bytes; ordered [nt][kt] so
// the GEMM K-loop streams contiguously.  Lane L dword j holds
// { W[k][n], W[k+1][n] } with n = nt*16 + (L&15),
// k = kt*32 + (j>>2)*16 + (L>>4)*8 + 2*(j&3).
__global__ __launch_bounds__(256) void pack_W_kernel(
    const float* __restrict__ W, unsigned int* __restrict__ Wp) {
    unsigned int tid = blockIdx.x * 256 + threadIdx.x;     // 4194304 dwords
    int j    = tid & 7;
    int lane = (tid >> 3) & 31;
    int kt   = (tid >> 8) & 63;
    int nt   = tid >> 14;
    int n = (nt << 4) + (lane & 15);
    int k = (kt << 5) + ((j >> 2) << 4) + ((lane >> 4) << 3) + ((j & 3) << 1);
    unsigned int lo = f32_to_bf16(W[(size_t)k * N4H + n]);
    unsigned int hi = f32_to_bf16(W[((size_t)k + 1) * N4H + n]);
    Wp[tid] = lo | (hi << 16);
}

__global__ __launch_bounds__(256) void init_state_kernel(
    float* __restrict__ c, unsigned short* __restrict__ hb) {
    int tid = blockIdx.x * 256 + threadIdx.x;              // B*H = 65536
    c[tid]  = 0.0f;
    hb[tid] = 0;
}

// ---------------------------------------------------------------------------
// Per-step GEMM: z[64,4096] = [x_t, h] @ W + b   (A,B bf16, C f32)
// 128 blocks x 8 waves.  Wave w: kh = w>>2 selects K-half (0: x, 1: h);
// sub = w&3 -> (mpair, ntl): M strip = (sub&1)*32 (two 16-row tiles),
// N tile = blockIdx.x*2 + (sub>>1).  kh=1 partials reduced via LDS.
// ---------------------------------------------------------------------------
__global__ __launch_bounds__(256) void lstm_gemm_step(
    const unsigned short* __restrict__ Wp,   // packed bf16 fragments
    const unsigned short* __restrict__ xb,   // [64][128][1024] bf16
    const unsigned short* __restrict__ hb,   // [64][1024] bf16
    const float* __restrict__ bias,          // [4096]
    float* __restrict__ z,                   // [64][4096] f32
    int t)
{
    __shared__ float red[4][2][32][8];                  // 8 KB partial sums

    const int lane   = threadIdx.x & 31;
    const int wave   = threadIdx.x >> 5;
    const int kh     = wave >> 2;                       // 0: x-half, 1: h-half
    const int sub    = wave & 3;
    const int m_base = (sub & 1) << 5;                  // 0 or 32
    const int nt     = (blockIdx.x << 1) + (sub >> 1);  // 0..255
    const int g      = lane >> 4;
    const int l16    = lane & 15;

    v8f acc0 = {};
    v8f acc1 = {};

    // packed-W stream: fragment (nt, kt) at (nt*64 + kt)*512 ushorts;
    // kh=1 starts at kt=32 (K = 1024..2047)
    const unsigned short* bp =
        Wp + ((size_t)nt << 15) + ((size_t)kh << 14) + lane * 16;

    const int m0 = m_base + l16;
    const unsigned short *a0, *a1;
    if (kh == 0) {                                      // A = x_t rows
        a0 = xb + (((size_t)m0 * T_ + t) << 10) + (g << 3);
        a1 = a0 + (((size_t)16 * T_) << 10);
    } else {                                            // A = h rows
        a0 = hb + ((size_t)m0 << 10) + (g << 3);
        a1 = a0 + (16 << 10);
    }

    #pragma unroll 4
    for (int kt = 0; kt < 32; ++kt) {
        __builtin_prefetch(bp + 512 * 8, 0, 1);         // global_prefetch_b8
        v16bf Bv = load_bfrag(bp);
        v16bf A0 = load_afrag(a0);
        v16bf A1 = load_afrag(a1);
        acc0 = __builtin_amdgcn_wmma_f32_16x16x32_bf16(
                   false, A0, false, Bv, (short)0, acc0, false, false);
        acc1 = __builtin_amdgcn_wmma_f32_16x16x32_bf16(
                   false, A1, false, Bv, (short)0, acc1, false, false);
        bp += 512; a0 += 32; a1 += 32;
    }

    // ---- cross-wave K reduction through LDS ----
    if (kh == 1) {
        union { v8f v; float4 q[2]; } u0, u1;
        u0.v = acc0; u1.v = acc1;
        float4* r0 = reinterpret_cast<float4*>(&red[sub][0][lane][0]);
        float4* r1 = reinterpret_cast<float4*>(&red[sub][1][lane][0]);
        r0[0] = u0.q[0]; r0[1] = u0.q[1];               // ds_store_b128
        r1[0] = u1.q[0]; r1[1] = u1.q[1];
    }
    __syncthreads();
    if (kh == 0) {
        const float4* r0 =
            reinterpret_cast<const float4*>(&red[sub][0][lane][0]);
        const float4* r1 =
            reinterpret_cast<const float4*>(&red[sub][1][lane][0]);
        union { v8f v; float4 q[2]; } p0, p1;
        p0.q[0] = r0[0]; p0.q[1] = r0[1];               // ds_load_b128
        p1.q[0] = r1[0]; p1.q[1] = r1[1];

        // C/D layout: acc elem j -> row = tile_base + j + (lane>=16 ? 8 : 0),
        // col = nt*16 + l16
        const int   n    = (nt << 4) + l16;
        const float bn   = bias[n];
        const int   row0 = m_base + (g << 3);
        #pragma unroll
        for (int j = 0; j < 8; ++j) {
            z[(size_t)(row0 + j)      * N4H + n] = acc0[j] + p0.v[j] + bn;
            z[(size_t)(row0 + 16 + j) * N4H + n] = acc1[j] + p1.v[j] + bn;
        }
    }
}

// ---------------------------------------------------------------------------
// Per-step gates: c' = c*sig(f+1) + sig(i)*tanh(j); h' = tanh(c')*sig(o)
// ---------------------------------------------------------------------------
__global__ __launch_bounds__(256) void lstm_gates_step(
    const float* __restrict__ z,        // [64][4096]
    float* __restrict__ c,              // [64][1024]
    unsigned short* __restrict__ hb,    // [64][1024] bf16 recurrent state
    float* __restrict__ y,              // [64][128][1024] output
    int t)
{
    int tid = blockIdx.x * 256 + threadIdx.x;   // 0 .. 65535
    int b = tid >> 10;
    int n = tid & 1023;
    const float* zr = z + ((size_t)b << 12);
    float zi = zr[n];
    float zj = zr[n + 1024];
    float zf = zr[n + 2048];
    float zo = zr[n + 3072];

    float si = 1.0f / (1.0f + __expf(-zi));
    float sf = 1.0f / (1.0f + __expf(-(zf + 1.0f)));   // FORGET_BIAS = 1.0
    float so = 1.0f / (1.0f + __expf(-zo));
    float tj = tanhf(zj);

    float nc = c[tid] * sf + si * tj;
    float nh = tanhf(nc) * so;

    c[tid]  = nc;
    hb[tid] = f32_to_bf16(nh);
    y[((size_t)b * T_ + t) * H_ + n] = nh;
}

// ---------------------------------------------------------------------------
// Host-side orchestration (graph-capture safe: only kernel launches on stream)
// ---------------------------------------------------------------------------
extern "C" void kernel_launch(void* const* d_in, const int* in_sizes, int n_in,
                              void* d_out, int out_size, void* d_ws, size_t ws_size,
                              hipStream_t stream) {
    (void)in_sizes; (void)n_in; (void)out_size; (void)ws_size;

    const float* x    = (const float*)d_in[0];   // [64,128,1024]
    const float* W    = (const float*)d_in[1];   // [2048,4096]
    const float* bias = (const float*)d_in[2];   // [4096]
    float*       y    = (float*)d_out;           // [64,128,1024]

    char* ws = (char*)d_ws;
    unsigned short* Wp = (unsigned short*)(ws);                              // 16 MB
    unsigned short* xb = (unsigned short*)(ws + (16u << 20));                // 16 MB
    unsigned short* hb = (unsigned short*)(ws + (32u << 20));                // 128 KB
    float*          c  = (float*)(ws + (32u << 20) + (128u << 10));          // 256 KB
    float*          z  = (float*)(ws + (32u << 20) + (384u << 10));          // 1 MB

    convert_x_kernel <<<32768, 256, 0, stream>>>(x, xb);
    pack_W_kernel    <<<16384, 256, 0, stream>>>(W, (unsigned int*)Wp);
    init_state_kernel<<<  256, 256, 0, stream>>>(c, hb);

    for (int t = 0; t < T_; ++t) {
        lstm_gemm_step <<<128, 256, 0, stream>>>(Wp, xb, hb, bias, z, t);
        lstm_gates_step<<<256, 256, 0, stream>>>(z, c, hb, y, t);
    }
}
